// BNReLULinear_24197845745956
// MI455X (gfx1250) — matrix operand order repaired and checked by reading, not compiled, not run on hardware
//
#include <hip/hip_runtime.h>

// ---------------------------------------------------------------------------
// BN (batch stats) + Linear(256->256, no bias) + ELU, N = 1,048,576 rows.
// Strategy (MI455X / gfx1250):
//   * memory-bound (~3.2 GB min traffic @ 23.3 TB/s ~= 140 us floor)
//   * bf16 WMMA (v_wmma_f32_16x16x32_bf16) with fp32 accumulation
//   * W staged per workgroup into 320 KB WGP LDS (pitch 264 kills conflicts)
//   * branchless ELU via hardware v_exp_f32 (TRANS pipe, co-executes)
//   * non-temporal x loads / out stores keep bf16 W hot in the 192 MB L2
// ---------------------------------------------------------------------------

#define F_DIM      256
#define ROWS       256                      // rows per workgroup tile
#define PITCH      264                      // bf16 elements per padded LDS row
#define N_TOTAL    1048576
#define STAT_ROWS  1024                     // rows per stats block
#define SMEM_BYTES ((F_DIM * PITCH + ROWS * PITCH) * 2 + 2 * F_DIM * 4)

typedef __attribute__((ext_vector_type(16))) __bf16         v16bf;
typedef __attribute__((ext_vector_type(8)))  float          v8f;
typedef __attribute__((ext_vector_type(4)))  float          v4f;
typedef __attribute__((ext_vector_type(4)))  unsigned int   v4u;
typedef __attribute__((ext_vector_type(4)))  unsigned short v4h;

union Frag {
    v16bf v;
    v4u   u[2];
};

__device__ __forceinline__ unsigned short f2bf(float f) {
    // round-to-nearest-even fp32 -> bf16
    unsigned int u = __float_as_uint(f);
    u += 0x7FFFu + ((u >> 16) & 1u);
    return (unsigned short)(u >> 16);
}

// --- kernel 0: zero the stats accumulators in workspace ---------------------
__global__ void bn_zero_ws_kernel(float* p, int n) {
    int i = blockIdx.x * blockDim.x + threadIdx.x;
    if (i < n) p[i] = 0.0f;
}

// --- kernel 1: per-feature sum / sum-of-squares -----------------------------
__global__ __launch_bounds__(256) void bn_col_stats_kernel(
    const float* __restrict__ x, float* __restrict__ sum, float* __restrict__ sumsq) {
    const int c = threadIdx.x;                       // one feature per lane
    const size_t r0 = (size_t)blockIdx.x * STAT_ROWS;
    const float* p = x + r0 * F_DIM + c;
    float s = 0.0f, q = 0.0f;
    for (int i = 0; i < STAT_ROWS; ++i) {
        float v = __builtin_nontemporal_load(p + (size_t)i * F_DIM);  // streaming
        s += v;
        q += v * v;
    }
    atomicAdd(&sum[c], s);                           // global_atomic_add_f32
    atomicAdd(&sumsq[c], q);
}

// --- kernel 2: fold mean/var/gamma/beta into scale/bias ---------------------
__global__ void bn_finalize_stats_kernel(
    const float* __restrict__ sum, const float* __restrict__ sumsq,
    const float* __restrict__ gamma, const float* __restrict__ beta,
    float* __restrict__ scale, float* __restrict__ bias) {
    int c = threadIdx.x;
    const float inv = 1.0f / (float)N_TOTAL;
    float m   = sum[c] * inv;
    float var = fmaxf(sumsq[c] * inv - m * m, 0.0f);  // biased variance
    float sc  = gamma[c] * rsqrtf(var + 1e-5f);
    scale[c]  = sc;
    bias[c]   = beta[c] - m * sc;
}

// --- kernel 3: W fp32 -> bf16 (done once; 128 KB stays L2-resident) ---------
__global__ __launch_bounds__(256) void bn_convert_w_kernel(
    const float* __restrict__ W, unsigned short* __restrict__ wbf) {
    int i = blockIdx.x * blockDim.x + threadIdx.x;   // float4 index
    v4f v = ((const v4f*)W)[i];
    v4h h;
    h.x = f2bf(v.x); h.y = f2bf(v.y); h.z = f2bf(v.z); h.w = f2bf(v.w);
    ((v4h*)wbf)[i] = h;
}

// --- kernel 4: fused normalize + bf16 WMMA GEMM + ELU -----------------------
__global__ __launch_bounds__(256) void bn_gemm_elu_kernel(
    const float* __restrict__ x, const float* __restrict__ scaleG,
    const float* __restrict__ biasG, const unsigned short* __restrict__ wbf,
    float* __restrict__ out) {
    extern __shared__ __align__(16) unsigned char smem[];
    unsigned short* sW = (unsigned short*)smem;        // [256][PITCH] bf16 W
    unsigned short* sX = sW + F_DIM * PITCH;           // [ROWS][PITCH] bf16 xn
    float* sScale = (float*)(sX + ROWS * PITCH);       // [256]
    float* sBias  = sScale + F_DIM;                    // [256]

    const int tid = threadIdx.x;
    const size_t rowBase = (size_t)blockIdx.x * ROWS;

    // stage per-feature scale/bias (blockDim == F_DIM == 256)
    sScale[tid] = scaleG[tid];
    sBias[tid]  = biasG[tid];

    // stage W (bf16) into padded LDS rows: 16 B per iteration per thread
    for (int e = tid; e < F_DIM * F_DIM / 8; e += 256) {
        int r = e >> 5;                // row
        int c = (e << 3) & 255;        // column
        *(v4u*)(sW + r * PITCH + c) = *(const v4u*)(wbf + r * F_DIM + c);
    }
    __syncthreads();                   // sScale/sBias ready

    // normalize the 256-row x tile, cast to bf16, store to LDS
    for (int e = tid; e < ROWS * F_DIM / 4; e += 256) {
        int r = e >> 6;
        int c = (e << 2) & 255;
        v4f xv = __builtin_nontemporal_load(
            (const v4f*)(x + (rowBase + r) * F_DIM + c));        // streaming
        v4f sc = *(const v4f*)(sScale + c);
        v4f bs = *(const v4f*)(sBias + c);
        v4h h;
        h.x = f2bf(fmaf(xv.x, sc.x, bs.x));
        h.y = f2bf(fmaf(xv.y, sc.y, bs.y));
        h.z = f2bf(fmaf(xv.z, sc.z, bs.z));
        h.w = f2bf(fmaf(xv.w, sc.w, bs.w));
        *(v4h*)(sX + r * PITCH + c) = h;
    }
    __syncthreads();                   // sW + sX ready

    // --- WMMA: each of 8 waves owns two 16-row strips x 16 col-tiles --------
    const int wave = tid >> 5;
    const int lane = tid & 31;
    const int lm   = lane & 15;        // row (A) / col (B,D) within tile
    const int g    = lane >> 4;        // half-wave selector

    for (int strip = 0; strip < 2; ++strip) {
        const int rowOff = wave * 32 + strip * 16;

        // A fragments for all 8 K-slices, kept in registers across col loop.
        // ISA A-layout (16-bit 16x32): lane lm holds K = g*8..g*8+7 (v0..3)
        // and K = 16+g*8..16+g*8+7 (v4..7)  => two 16 B chunks, 16 elems apart.
        Frag a[8];
        const unsigned short* ap = sX + (rowOff + lm) * PITCH + g * 8;
#pragma unroll
        for (int kb = 0; kb < 8; ++kb) {
            const unsigned short* p = ap + kb * 32;
            a[kb].u[0] = *(const v4u*)(p);
            a[kb].u[1] = *(const v4u*)(p + 16);
        }

#pragma unroll
        for (int cb = 0; cb < 16; ++cb) {
            v8f acc = {};
            // B-layout (16-bit 32x16): lane lm = output col n, halves g hold
            // K = g*8..g*8+7 and 16+g*8..16+g*8+7; B = W^T so col n of B = row n of W.
            const unsigned short* bp = sW + (cb * 16 + lm) * PITCH + g * 8;
#pragma unroll
            for (int kb = 0; kb < 8; ++kb) {
                Frag b;
                const unsigned short* p = bp + kb * 32;
                b.u[0] = *(const v4u*)(p);
                b.u[1] = *(const v4u*)(p + 16);
                acc = __builtin_amdgcn_wmma_f32_16x16x32_bf16(
                    false, a[kb].v, false, b.v, (short)0, acc, false, false);
            }
            // C/D layout: vgpr j -> row (rowOff + 8*g + j), lane lm -> col n.
            const int n = cb * 16 + lm;
            const size_t outRow0 = rowBase + rowOff + g * 8;
#pragma unroll
            for (int j = 0; j < 8; ++j) {
                float y = acc[j];
                // branchless ELU(alpha=1): hardware v_exp_f32 + v_cndmask,
                // no divergent s_cbranch per element
                float e = __expf(y) - 1.0f;
                y = (y > 0.0f) ? y : e;
                __builtin_nontemporal_store(
                    y, out + (outRow0 + (size_t)j) * F_DIM + n);  // streaming
            }
        }
    }
}

// ---------------------------------------------------------------------------
extern "C" void kernel_launch(void* const* d_in, const int* in_sizes, int n_in,
                              void* d_out, int out_size, void* d_ws, size_t ws_size,
                              hipStream_t stream) {
    const float* x     = (const float*)d_in[0];
    const float* gamma = (const float*)d_in[1];
    const float* beta  = (const float*)d_in[2];
    const float* W     = (const float*)d_in[3];
    float* out = (float*)d_out;

    // workspace layout (floats): [0,256) sum | [256,512) sumsq |
    // [512,768) scale | [768,1024) bias | then 256x256 bf16 W (128 KB)
    float* wsf = (float*)d_ws;
    float* sum   = wsf;
    float* sumsq = wsf + 256;
    float* scale = wsf + 512;
    float* bias  = wsf + 768;
    unsigned short* wbf = (unsigned short*)(wsf + 1024);

    // allow >64 KB dynamic LDS (gfx1250 WGP has 320 KB)
    (void)hipFuncSetAttribute((const void*)bn_gemm_elu_kernel,
                              hipFuncAttributeMaxDynamicSharedMemorySize,
                              (int)SMEM_BYTES);

    bn_zero_ws_kernel<<<2, 256, 0, stream>>>(wsf, 512);
    bn_col_stats_kernel<<<N_TOTAL / STAT_ROWS, 256, 0, stream>>>(x, sum, sumsq);
    bn_finalize_stats_kernel<<<1, 256, 0, stream>>>(sum, sumsq, gamma, beta, scale, bias);
    bn_convert_w_kernel<<<F_DIM * F_DIM / 4 / 256, 256, 0, stream>>>(W, wbf);
    bn_gemm_elu_kernel<<<N_TOTAL / ROWS, 256, SMEM_BYTES, stream>>>(x, scale, bias, wbf, out);
}
